// RetrievalCrossAttention_40200893890570
// MI455X (gfx1250) — matrix operand order repaired and checked by reading, not compiled, not run on hardware
//
#include <hip/hip_runtime.h>
#include <hip/hip_bf16.h>
#include <math.h>

// ---------------- problem dims ----------------
#define B_ROWS 16384
#define KRET   32
#define QD     512
#define HDIM   256
#define NH     4
#define HD     64
#define NCODE  21      // NR+1
#define KIN    45
#define TM     16      // batch rows per block
#define NTILE  16      // 256 / 16 N-tiles
#define NPW    4       // N-tiles per wave (4 waves * 4 = 16)
#define THREADS 128

typedef __attribute__((ext_vector_type(16))) _Float16 v16h;
typedef __attribute__((ext_vector_type(8)))  _Float16 v8h;
typedef __attribute__((ext_vector_type(8)))  float    v8f;

union AFrag { v16h v; v8h h[2]; };

// ---- workspace layout (bytes); total = 888832 ----
#define WS_BQ    0          // f16 512x256 swizzled           262144
#define WS_BOA   262144     // f16 256x256 swizzled           131072
#define WS_BM1   393216     // f16 256x256 swizzled           131072
#define WS_BM2   524288     // f16 256x256 swizzled           131072
#define WS_B2    655360     // f16 160x256 swizzled (attn)     81920
#define WS_WKF   737280     // f32 256x45                      46080
#define WS_WVF   783360     // f32 256x45                      46080
#define WS_EK    829440     // f32 21x256                      21504
#define WS_EV    850944     // f32 21x256                      21504
#define WS_WG    872448     // f32 13x256                      13312
#define WS_BQF   885760     // f32 256
#define WS_BKF   886784     // f32 256
#define WS_BVF   887808     // f32 256

// store element W[k][n] into pre-swizzled WMMA B layout:
// lane = (n&15) + (k%32>=16 ? 16:0), half = k%32 mod 16, tile-major (kt,nt)
__device__ __forceinline__ void sw_store(_Float16* Bsw, int k, int n, float v) {
  int kt = k >> 5, kl = k & 31;
  int nt = n >> 4;
  int ln = (n & 15) + ((kl & 16) ? 16 : 0);
  Bsw[(((((kt << 4) + nt) << 5) + ln) << 4) + (kl & 15)] = (_Float16)v;
}

__device__ __forceinline__ void zacc(v8f acc[NPW]) {
  v8f z = {0.f, 0.f, 0.f, 0.f, 0.f, 0.f, 0.f, 0.f};
#pragma unroll
  for (int i = 0; i < NPW; ++i) acc[i] = z;
}

// C(16xN) += A(16x32*KT, LDS, f16) * B(pre-swizzled, global, f16)
template<int KT>
__device__ __forceinline__ void wmma_gemm(const _Float16* As, int astride,
                                          const _Float16* Bsw,
                                          int wave, int lane, v8f acc[NPW]) {
#pragma unroll
  for (int kt = 0; kt < KT; ++kt) {
    // A fragment, contiguous K to match swizzled B: lanes 0-15 K=0..15, lanes 16-31 K=16..31
    const _Float16* ap = As + (lane & 15) * astride + kt * 32 + ((lane & 16) ? 16 : 0);
    AFrag fa;
    fa.h[0] = *(const v8h*)ap;
    fa.h[1] = *(const v8h*)(ap + 8);
#pragma unroll
    for (int i = 0; i < NPW; ++i) {
      int nt = wave * NPW + i;
      v16h bf = *(const v16h*)(Bsw + ((((kt * NTILE + nt) << 5) + lane) << 4));
      acc[i] = __builtin_amdgcn_wmma_f32_16x16x32_f16(false, fa.v, false, bf,
                                                      (short)0, acc[i], false, false);
    }
  }
}

// writeback C tiles -> f16 LDS. MODE 0: +bias, 1: +bias then exact gelu, 2: raw
template<int MODE>
__device__ __forceinline__ void wb_f16(v8f acc[NPW], const float* bias,
                                       _Float16* dst, int wave, int lane) {
#pragma unroll
  for (int i = 0; i < NPW; ++i) {
    int n = ((wave * NPW + i) << 4) + (lane & 15);
#pragma unroll
    for (int r = 0; r < 8; ++r) {
      int m = r + ((lane & 16) ? 8 : 0);
      float v = acc[i][r];
      if (MODE != 2) v += bias[n];
      if (MODE == 1) v = 0.5f * v * (1.0f + erff(v * 0.70710678118f));
      dst[m * HDIM + n] = (_Float16)v;
    }
  }
}

// ================= prep kernels (run once per launch, tiny) =================

// W_qf = W_qa @ W_qproj  (256x512) -> swizzled f16
__global__ void prep_qf(const float* __restrict__ Wqa, const float* __restrict__ Wqp,
                        _Float16* __restrict__ Bq) {
  int gid = blockIdx.x * 256 + threadIdx.x;   // 131072
  int n = gid >> 9, k = gid & 511;
  float s = 0.f;
  for (int i = 0; i < HDIM; ++i) s += Wqa[n * HDIM + i] * Wqp[i * QD + k];
  sw_store(Bq, k, n, s);
}

// W_kf = W_ka @ W_kproj, W_vf = W_va @ W_vproj  (256x45 each, f32)
__global__ void prep_kv(const float* __restrict__ Wka, const float* __restrict__ Wkp,
                        const float* __restrict__ Wva, const float* __restrict__ Wvp,
                        float* __restrict__ Wkf, float* __restrict__ Wvf) {
  int gid = blockIdx.x * 256 + threadIdx.x;   // 23040
  int which = gid / (HDIM * KIN);
  int rem = gid % (HDIM * KIN);
  int n = rem / KIN, j = rem % KIN;
  const float* WA = which ? Wva : Wka;
  const float* WP = which ? Wvp : Wkp;
  float s = 0.f;
  for (int i = 0; i < HDIM; ++i) s += WA[n * HDIM + i] * WP[i * KIN + j];
  (which ? Wvf : Wkf)[n * KIN + j] = s;
}

// fused biases
__global__ void prep_bias(const float* Wqa, const float* bqp, const float* bqa,
                          const float* Wka, const float* bkp, const float* bka,
                          const float* Wva, const float* bvp, const float* bva,
                          float* bqf, float* bkf, float* bvf) {
  int n = threadIdx.x;
  float sq = bqa[n], sk = bka[n], sv = bva[n];
  for (int i = 0; i < HDIM; ++i) {
    sq += Wqa[n * HDIM + i] * bqp[i];
    sk += Wka[n * HDIM + i] * bkp[i];
    sv += Wva[n * HDIM + i] * bvp[i];
  }
  bqf[n] = sq; bkf[n] = sk; bvf[n] = sv;
}

// relayout W_oa / W_m1 / W_m2 into swizzled f16 B (B[k][n] = W[n][k])
__global__ void prep_sw(const float* Woa, const float* Wm1, const float* Wm2,
                        _Float16* Boa, _Float16* Bm1, _Float16* Bm2) {
  int gid = blockIdx.x * 256 + threadIdx.x;   // 196608
  int sel = gid >> 16, rem = gid & 65535;
  int n = rem >> 8, k = rem & 255;
  const float* W = sel == 0 ? Woa : (sel == 1 ? Wm1 : Wm2);
  _Float16* Bsw = sel == 0 ? Boa : (sel == 1 ? Bm1 : Bm2);
  sw_store(Bsw, k, n, W[n * HDIM + k]);
}

// Ek/Ev code tables (bias folded in) + Wg (13 fused key feature columns, transposed)
__global__ void prep_tab1(const float* emb, const float* Wkf, const float* Wvf,
                          const float* bkf, const float* bvf,
                          float* Ek, float* Ev, float* Wg) {
  int gid = blockIdx.x * 256 + threadIdx.x;   // 14080
  const int E = NCODE * HDIM;                  // 5376
  if (gid < 2 * E) {
    int which = gid / E, rem = gid % E;
    int c = rem >> 8, n = rem & 255;
    const float* W = which ? Wvf : Wkf;
    float s = (which ? bvf : bkf)[n];
    for (int e = 0; e < 32; ++e) s += emb[c * 32 + e] * W[n * KIN + 12 + e];
    (which ? Ev : Ek)[c * HDIM + n] = s;
  } else {
    int rem = gid - 2 * E;                     // 13*256
    int j = rem >> 8, n = rem & 255;
    Wg[j * HDIM + n] = Wkf[n * KIN + (j < 12 ? j : 44)];
  }
}

// B2: block-diagonal per-head 160x256 attention-combine matrix, swizzled f16
__global__ void prep_tab2(const float* Wvf, const float* Ev, _Float16* B2) {
  int gid = blockIdx.x * 256 + threadIdx.x;   // 40960
  int col = gid >> 8, n = gid & 255;
  int h = n >> 6;
  float v = 0.f;
  if (col < NH * 13) {                         // pf part
    int hh = col / 13, j = col % 13;
    if (hh == h) v = Wvf[n * KIN + (j < 12 ? j : 44)];
  } else if (col < NH * 13 + NH * NCODE) {     // code-mass part
    int cc = col - NH * 13;
    int hh = cc / NCODE, c = cc % NCODE;
    if (hh == h) v = Ev[c * HDIM + n];
  }                                            // cols 136..159 stay zero
  sw_store(B2, col, n, v);
}

// ================= main fused kernel =================

__global__ __launch_bounds__(THREADS)
void rca_main(const float* __restrict__ query,
              const float* __restrict__ shifts,
              const unsigned char* __restrict__ masks,
              const int* __restrict__ codes,
              const float* __restrict__ dist,
              const unsigned char* __restrict__ valid,
              const _Float16* __restrict__ Bq, const _Float16* __restrict__ Boa,
              const _Float16* __restrict__ Bm1, const _Float16* __restrict__ Bm2,
              const _Float16* __restrict__ B2,
              const float* __restrict__ bqf, const float* __restrict__ boa,
              const float* __restrict__ bm1, const float* __restrict__ bm2,
              const float* __restrict__ Ek, const float* __restrict__ Wg,
              const float* __restrict__ fallback,
              float* __restrict__ out) {
  __shared__ alignas(16) _Float16 Ash[TM * QD];          // 16 KB: gemm1 A; later buf0/buf1
  __shared__ alignas(16) _Float16 qs[TM * HDIM];         // 8 KB fused q, f16
  __shared__ _Float16 feat[TM * KRET * 13];              // 13.3 KB kv features
  __shared__ float sc[TM * NH * KRET];                   // 8 KB scores -> probs
  __shared__ float gt[TM * NH * (NCODE + 13)];           // per (m,h): t[21] | g[13]
  __shared__ unsigned char cds[TM * KRET];
  __shared__ unsigned int vmask[TM];

  const int tid = threadIdx.x, lane = tid & 31, wave = tid >> 5;
  const int b0 = blockIdx.x * TM;
  _Float16* buf0 = Ash;
  _Float16* buf1 = Ash + TM * HDIM;

  if (tid < TM) vmask[tid] = 0u;
  __syncthreads();

  // stage query tile (f32 -> f16, row-major A)
  for (int i = tid; i < TM * QD; i += THREADS) {
    int m = i >> 9, k = i & (QD - 1);
    Ash[i] = (_Float16)query[(b0 + m) * QD + k];
  }
  // stage retrieval features
  for (int i = tid; i < TM * KRET; i += THREADS) {
    int m = i >> 5, k = i & 31;
    int g = (b0 + m) * KRET + k;
    _Float16* f = &feat[i * 13];
#pragma unroll
    for (int j = 0; j < 6; ++j) f[j] = (_Float16)shifts[g * 6 + j];
#pragma unroll
    for (int j = 0; j < 6; ++j) f[6 + j] = (_Float16)(masks[g * 6 + j] ? 1.0f : 0.0f);
    f[12] = (_Float16)dist[g];
    cds[i] = (unsigned char)codes[g];
    if (valid[g]) atomicOr(&vmask[m], 1u << k);
  }
  __syncthreads();

  // GEMM1: q(16x256) = queryTile(16x512) @ W_qf^T  [16 K-steps]
  {
    v8f acc[NPW]; zacc(acc);
    wmma_gemm<QD / 32>(Ash, QD, Bq, wave, lane, acc);
    wb_f16<0>(acc, bqf, qs, wave, lane);
  }
  __syncthreads();

  // g[m][h][j] = q_h . Wg_j ; t[m][h][c] = q_h . Ek_c   (VALU, small)
  for (int i = tid; i < TM * NH * 13; i += THREADS) {
    int m = i / (NH * 13), rem = i % (NH * 13), h = rem / 13, j = rem % 13;
    const _Float16* qr = &qs[m * HDIM + h * HD];
    const float* w = &Wg[j * HDIM + h * HD];
    float s = 0.f;
    for (int u = 0; u < HD; ++u) s += (float)qr[u] * w[u];
    gt[(m * NH + h) * (NCODE + 13) + NCODE + j] = s;
  }
  for (int i = tid; i < TM * NH * NCODE; i += THREADS) {
    int m = i / (NH * NCODE), rem = i % (NH * NCODE), h = rem / NCODE, c = rem % NCODE;
    const _Float16* qr = &qs[m * HDIM + h * HD];
    const float* e = &Ek[c * HDIM + h * HD];
    float s = 0.f;
    for (int u = 0; u < HD; ++u) s += (float)qr[u] * e[u];
    gt[(m * NH + h) * (NCODE + 13) + c] = s;
  }
  __syncthreads();

  // scores + mask (slot-0 unmask when nothing valid, as in reference)
  for (int i = tid; i < TM * NH * KRET; i += THREADS) {
    int m = i >> 7, rem = i & 127, h = rem >> 5, k = rem & 31;
    const _Float16* f = &feat[(m * KRET + k) * 13];
    const float* gp = &gt[(m * NH + h) * (NCODE + 13)];
    float s = gp[cds[m * KRET + k]];             // q . Ek[code]  (bias folded)
#pragma unroll
    for (int j = 0; j < 13; ++j) s += (float)f[j] * gp[NCODE + j];
    s *= 0.125f;                                  // 1/sqrt(64)
    unsigned vm = vmask[m];
    bool keep = ((vm >> k) & 1u) || (vm == 0u && k == 0);
    sc[i] = keep ? s : -1e30f;
  }
  __syncthreads();

  // softmax over K per (m,h)
  if (tid < TM * NH) {
    float* row = &sc[tid * KRET];
    float mx = -1e30f;
    for (int k = 0; k < KRET; ++k) mx = fmaxf(mx, row[k]);
    float sum = 0.f;
    for (int k = 0; k < KRET; ++k) { float e = __expf(row[k] - mx); row[k] = e; sum += e; }
    float inv = 1.0f / sum;
    for (int k = 0; k < KRET; ++k) row[k] *= inv;
  }
  __syncthreads();

  // build combined A2 = [pf(4x13) | code-mass(4x21) | 0...] in buf0
  for (int i = tid; i < TM * HDIM; i += THREADS) buf0[i] = (_Float16)0.f;
  __syncthreads();
  for (int i = tid; i < TM * NH * 13; i += THREADS) {
    int m = i / (NH * 13), rem = i % (NH * 13), h = rem / 13, j = rem % 13;
    const float* p = &sc[(m * NH + h) * KRET];
    float s = 0.f;
    for (int k = 0; k < KRET; ++k) s += p[k] * (float)feat[(m * KRET + k) * 13 + j];
    buf0[m * HDIM + h * 13 + j] = (_Float16)s;
  }
  for (int i = tid; i < TM * NH * NCODE; i += THREADS) {
    int m = i / (NH * NCODE), rem = i % (NH * NCODE), h = rem / NCODE, c = rem % NCODE;
    const float* p = &sc[(m * NH + h) * KRET];
    const unsigned char* cc = &cds[m * KRET];
    float s = 0.f;
    for (int k = 0; k < KRET; ++k) s += (cc[k] == c) ? p[k] : 0.f;
    buf0[m * HDIM + NH * 13 + h * NCODE + c] = (_Float16)s;
  }
  __syncthreads();

  // attn = A2(16x160) @ B2   (value weights + code tables, biases folded)
  {
    v8f acc[NPW]; zacc(acc);
    wmma_gemm<5>(buf0, HDIM, B2, wave, lane, acc);
    wb_f16<2>(acc, nullptr, buf1, wave, lane);
  }
  __syncthreads();
  // attn_out = attn @ W_oa^T + b_oa
  {
    v8f acc[NPW]; zacc(acc);
    wmma_gemm<HDIM / 32>(buf1, HDIM, Boa, wave, lane, acc);
    wb_f16<0>(acc, boa, buf0, wave, lane);
  }
  __syncthreads();
  // h = gelu(attn_out @ W_m1^T + b_m1)    (exact erf gelu)
  {
    v8f acc[NPW]; zacc(acc);
    wmma_gemm<HDIM / 32>(buf0, HDIM, Bm1, wave, lane, acc);
    wb_f16<1>(acc, bm1, buf1, wave, lane);
  }
  __syncthreads();
  // out = h @ W_m2^T + b_m2 ; fallback where no valid retrieval
  {
    v8f acc[NPW]; zacc(acc);
    wmma_gemm<HDIM / 32>(buf1, HDIM, Bm2, wave, lane, acc);
#pragma unroll
    for (int i = 0; i < NPW; ++i) {
      int n = ((wave * NPW + i) << 4) + (lane & 15);
#pragma unroll
      for (int r = 0; r < 8; ++r) {
        int m = r + ((lane & 16) ? 8 : 0);
        float v = acc[i][r] + bm2[n];
        out[(b0 + m) * HDIM + n] = vmask[m] ? v : fallback[n];
      }
    }
  }
}

// ================= host launch =================

extern "C" void kernel_launch(void* const* d_in, const int* in_sizes, int n_in,
                              void* d_out, int out_size, void* d_ws, size_t ws_size,
                              hipStream_t stream) {
  const float* query   = (const float*)d_in[0];
  // d_in[1] query_residue_code: unused by the reference forward
  const float* shifts  = (const float*)d_in[2];
  const unsigned char* masks = (const unsigned char*)d_in[3];
  const int*   codes   = (const int*)d_in[4];
  const float* dist    = (const float*)d_in[5];
  const unsigned char* valid = (const unsigned char*)d_in[6];
  const float* Wqproj = (const float*)d_in[7];
  const float* bqproj = (const float*)d_in[8];
  const float* emb    = (const float*)d_in[9];
  const float* Wkproj = (const float*)d_in[10];
  const float* bkproj = (const float*)d_in[11];
  const float* Wvproj = (const float*)d_in[12];
  const float* bvproj = (const float*)d_in[13];
  const float* Wqa = (const float*)d_in[14];
  const float* Wka = (const float*)d_in[15];
  const float* Wva = (const float*)d_in[16];
  const float* bqa = (const float*)d_in[17];
  const float* bka = (const float*)d_in[18];
  const float* bva = (const float*)d_in[19];
  const float* Woa = (const float*)d_in[20];
  const float* boa = (const float*)d_in[21];
  const float* Wm1 = (const float*)d_in[22];
  const float* bm1 = (const float*)d_in[23];
  const float* Wm2 = (const float*)d_in[24];
  const float* bm2 = (const float*)d_in[25];
  const float* fallback = (const float*)d_in[26];

  char* ws = (char*)d_ws;
  _Float16* Bq  = (_Float16*)(ws + WS_BQ);
  _Float16* Boa = (_Float16*)(ws + WS_BOA);
  _Float16* Bm1 = (_Float16*)(ws + WS_BM1);
  _Float16* Bm2 = (_Float16*)(ws + WS_BM2);
  _Float16* B2  = (_Float16*)(ws + WS_B2);
  float* Wkf = (float*)(ws + WS_WKF);
  float* Wvf = (float*)(ws + WS_WVF);
  float* Ek  = (float*)(ws + WS_EK);
  float* Ev  = (float*)(ws + WS_EV);
  float* Wg  = (float*)(ws + WS_WG);
  float* bqf = (float*)(ws + WS_BQF);
  float* bkf = (float*)(ws + WS_BKF);
  float* bvf = (float*)(ws + WS_BVF);

  prep_qf  <<<512, 256, 0, stream>>>(Wqa, Wqproj, Bq);
  prep_kv  <<<90, 256, 0, stream>>>(Wka, Wkproj, Wva, Wvproj, Wkf, Wvf);
  prep_bias<<<1, 256, 0, stream>>>(Wqa, bqproj, bqa, Wka, bkproj, bka,
                                   Wva, bvproj, bva, bqf, bkf, bvf);
  prep_sw  <<<768, 256, 0, stream>>>(Woa, Wm1, Wm2, Boa, Bm1, Bm2);
  prep_tab1<<<55, 256, 0, stream>>>(emb, Wkf, Wvf, bkf, bvf, Ek, Ev, Wg);
  prep_tab2<<<160, 256, 0, stream>>>(Wvf, Ev, B2);

  rca_main<<<B_ROWS / TM, THREADS, 0, stream>>>(
      query, shifts, masks, codes, dist, valid,
      Bq, Boa, Bm1, Bm2, B2,
      bqf, boa, bm1, bm2, Ek, Wg, fallback,
      (float*)d_out);
}
